// TimeSeriesTransformer_35364760715921
// MI455X (gfx1250) — matrix operand order repaired and checked by reading, not compiled, not run on hardware
//
#include <hip/hip_runtime.h>
#include <hip/hip_bf16.h>
#include <math.h>

// ---------------------------------------------------------------------------
// Types for CDNA5 WMMA (wave32): v16bf A/B fragments, v8f C/D fragments
// ---------------------------------------------------------------------------
typedef __attribute__((ext_vector_type(16))) __bf16 v16bf;
typedef __attribute__((ext_vector_type(8)))  float  v8f;
typedef unsigned short u16;

union FragAB { v16bf v; u16 u[16]; uint4 q[2]; };
union Pack8 { uint4 q; u16 u[8]; };

__device__ __forceinline__ u16 f2bf(float f) {
    unsigned u = __float_as_uint(f);
    unsigned r = (u + 0x7FFFu + ((u >> 16) & 1u)) >> 16;   // RNE
    return (u16)r;
}

__device__ __forceinline__ v8f vzero8() {
    v8f z = {0.f, 0.f, 0.f, 0.f, 0.f, 0.f, 0.f, 0.f};
    return z;
}

// CDNA5 async global->LDS copy (ASYNCcnt tracked, no VGPR round-trip)
__device__ __forceinline__ void async_load_b128(void* lds_ptr, const void* gptr) {
    unsigned loff = (unsigned)(size_t)lds_ptr;      // low 32 bits = LDS offset
    asm volatile("global_load_async_to_lds_b128 %0, %1, off"
                 :: "v"(loff), "v"(gptr)
                 : "memory");
}
__device__ __forceinline__ void wait_async0() {
    asm volatile("s_wait_asynccnt 0x0" ::: "memory");
}

// Model constants
#define BD   2
#define SD   1024
#define DD   1024
#define HH   16
#define HDD  64
#define DFFD 4096
#define WIND 5
#define LD   6
#define MTOK (BD * SD)           // 2048 tokens

// ---------------------------------------------------------------------------
// Weight convert + transpose: src f32 [K][N] row-major -> dst bf16 [N][K]
// ---------------------------------------------------------------------------
#define TT 32
__global__ __launch_bounds__(256) void cvtT_kernel(const float* __restrict__ src,
                                                   u16* __restrict__ dst,
                                                   int K, int N) {
    __shared__ u16 tile[TT][TT + 2];
    const int bk = blockIdx.y * TT, bn = blockIdx.x * TT;
    const int tx = threadIdx.x & 31, ty = threadIdx.x >> 5;   // 32 x 8
#pragma unroll
    for (int r = ty; r < TT; r += 8)
        tile[r][tx] = f2bf(src[(size_t)(bk + r) * N + bn + tx]);
    __syncthreads();
#pragma unroll
    for (int r = ty; r < TT; r += 8)
        dst[(size_t)(bn + r) * K + bk + tx] = tile[tx][r];
}

// ---------------------------------------------------------------------------
// LayerNorm over D=1024, one block per token, bf16 output
// ---------------------------------------------------------------------------
__global__ __launch_bounds__(256) void ln_bf16_kernel(const float* __restrict__ x,
                                                      const float* __restrict__ w,
                                                      const float* __restrict__ b,
                                                      u16* __restrict__ out) {
    const int row = blockIdx.x;
    const int tid = threadIdx.x;
    const float* xr = x + (size_t)row * DD;
    float vals[4], s = 0.f, sq = 0.f;
#pragma unroll
    for (int k = 0; k < 4; k++) {
        float v = xr[tid + 256 * k];
        vals[k] = v; s += v; sq += v * v;
    }
#pragma unroll
    for (int m = 1; m < 32; m <<= 1) {
        s  += __shfl_xor(s,  m, 32);
        sq += __shfl_xor(sq, m, 32);
    }
    __shared__ float sm_s[8], sm_q[8];
    const int wid = tid >> 5, lane = tid & 31;
    if (lane == 0) { sm_s[wid] = s; sm_q[wid] = sq; }
    __syncthreads();
    float S = 0.f, Q = 0.f;
#pragma unroll
    for (int i = 0; i < 8; i++) { S += sm_s[i]; Q += sm_q[i]; }
    const float mean = S * (1.f / DD);
    const float var  = Q * (1.f / DD) - mean * mean;
    const float rstd = rsqrtf(var + 1e-5f);
#pragma unroll
    for (int k = 0; k < 4; k++) {
        int c = tid + 256 * k;
        out[(size_t)row * DD + c] = f2bf((vals[k] - mean) * rstd * w[c] + b[c]);
    }
}

// ---------------------------------------------------------------------------
// bf16 GEMM: C[M,N] = act( A[M,K] @ Bt^T + bias (+ residual) )
//   A : [M][K] bf16 row-major,  Bt : [N][K] bf16 (pre-transposed weights)
// 128x128 block tile, 8 waves, 2x4 WMMA per wave.
// Double-buffered LDS fed by global_load_async_to_lds_b128.
// ---------------------------------------------------------------------------
#define TM 128
#define TN 128
#define TK 32

__global__ __launch_bounds__(256) void gemm_bf16_kernel(
    const u16* __restrict__ A, const u16* __restrict__ Bt,
    const float* __restrict__ bias, const float* __restrict__ residual,
    float* __restrict__ Cf, u16* __restrict__ Cb,
    int M, int N, int K, int act)
{
    __shared__ u16 As[2][TM][TK + 8];     // row stride 80B (16B aligned)
    __shared__ u16 Bs[2][TN][TK + 8];

    const int bm = blockIdx.y * TM;
    const int bn = blockIdx.x * TN;
    const int tid = threadIdx.x;
    const int wid = tid >> 5, lane = tid & 31;
    const int half = lane >> 4, l16 = lane & 15;
    const int wm = (wid >> 1) * 32;    // 0,32,64,96
    const int wn = (wid & 1) * 64;     // 0,64

    // Static chunk assignment: thread handles chunks {tid, tid+256} of 512
    const int r0 = tid >> 2,          k0c = (tid & 3) * 8;
    const int r1 = (tid + 256) >> 2,  k1c = k0c;            // (tid+256)&3 == tid&3

    v8f acc[2][4];
#pragma unroll
    for (int i = 0; i < 2; i++)
#pragma unroll
        for (int j = 0; j < 4; j++) acc[i][j] = vzero8();

    // Prologue: stream first K-tile into buffer 0
    async_load_b128(&As[0][r0][k0c], &A[(size_t)(bm + r0) * K + k0c]);
    async_load_b128(&As[0][r1][k1c], &A[(size_t)(bm + r1) * K + k1c]);
    async_load_b128(&Bs[0][r0][k0c], &Bt[(size_t)(bn + r0) * K + k0c]);
    async_load_b128(&Bs[0][r1][k1c], &Bt[(size_t)(bn + r1) * K + k1c]);

    const int nk = K / TK;
    for (int it = 0; it < nk; it++) {
        wait_async0();
        __syncthreads();
        const int buf = it & 1;
        if (it + 1 < nk) {                 // stream next tile into other buffer
            const int nxt = buf ^ 1;
            const int kt = (it + 1) * TK;
            async_load_b128(&As[nxt][r0][k0c], &A[(size_t)(bm + r0) * K + kt + k0c]);
            async_load_b128(&As[nxt][r1][k1c], &A[(size_t)(bm + r1) * K + kt + k1c]);
            async_load_b128(&Bs[nxt][r0][k0c], &Bt[(size_t)(bn + r0) * K + kt + k0c]);
            async_load_b128(&Bs[nxt][r1][k1c], &Bt[(size_t)(bn + r1) * K + kt + k1c]);
        }

        FragAB ua[2], ub[4];
#pragma unroll
        for (int i = 0; i < 2; i++) {
            const int row = wm + 16 * i + l16;
            ua[i].q[0] = *(const uint4*)&As[buf][row][half * 8];
            ua[i].q[1] = *(const uint4*)&As[buf][row][16 + half * 8];
        }
#pragma unroll
        for (int j = 0; j < 4; j++) {
            const int col = wn + 16 * j + l16;
            ub[j].q[0] = *(const uint4*)&Bs[buf][col][half * 8];
            ub[j].q[1] = *(const uint4*)&Bs[buf][col][16 + half * 8];
        }
#pragma unroll
        for (int i = 0; i < 2; i++)
#pragma unroll
            for (int j = 0; j < 4; j++)
                acc[i][j] = __builtin_amdgcn_wmma_f32_16x16x32_bf16(
                    false, ua[i].v, false, ub[j].v, (short)0, acc[i][j], false, false);
    }

    // Epilogue: bias + residual + exact GELU + f32/bf16 store
#pragma unroll
    for (int i = 0; i < 2; i++)
#pragma unroll
        for (int j = 0; j < 4; j++)
#pragma unroll
            for (int v = 0; v < 8; v++) {
                const int row = bm + wm + 16 * i + v + 8 * half;
                const int col = bn + wn + 16 * j + l16;
                float xv = acc[i][j][v] + bias[col];
                if (residual) xv += residual[(size_t)row * N + col];
                if (act) xv = 0.5f * xv * (1.0f + erff(xv * 0.70710678118654752f));
                if (Cf) Cf[(size_t)row * N + col] = xv;
                if (Cb) Cb[(size_t)row * N + col] = f2bf(xv);
            }
}

// ---------------------------------------------------------------------------
// Flash-style causal attention over bf16 QKV, one block per (q-tile, head, b).
// Q/K tiles streamed via async global->LDS; V transposed manually.
// ---------------------------------------------------------------------------
__global__ __launch_bounds__(128) void attn_kernel(const u16* __restrict__ qkv,
                                                   u16* __restrict__ out) {
    __shared__ u16 Qs[64][72], Ks[64][72], VsT[64][72];   // 144B rows (16B aligned)
    __shared__ u16 Ps[4][16][72];

    const int qt = blockIdx.x, h = blockIdx.y, b = blockIdx.z;
    const int qbase = qt * 64;
    const int tid = threadIdx.x;
    const int wid = tid >> 5, lane = tid & 31;
    const int half = lane >> 4, l16 = lane & 15;
    const float scale = 0.125f;                     // 1/sqrt(64)
    const size_t rowstride = 3 * DD;
    const u16* base = qkv + (size_t)b * SD * rowstride;

    // Q tile: 512 16B chunks, 4 per thread, async copy
#pragma unroll
    for (int c = tid; c < 512; c += 128) {
        const int r = c >> 3, dof = (c & 7) * 8;
        async_load_b128(&Qs[r][dof],
                        &base[(size_t)(qbase + r) * rowstride + h * HDD + dof]);
    }

    v8f oacc[4];
#pragma unroll
    for (int t = 0; t < 4; t++) oacc[t] = vzero8();
    float rm[8], rl[8];
#pragma unroll
    for (int v = 0; v < 8; v++) { rm[v] = -1e30f; rl[v] = 0.f; }

    for (int j = 0; j <= qt; j++) {
        const int kb = j * 64;
        // K tile async
#pragma unroll
        for (int c = tid; c < 512; c += 128) {
            const int r = c >> 3, dof = (c & 7) * 8;
            async_load_b128(&Ks[r][dof],
                            &base[(size_t)(kb + r) * rowstride + DD + h * HDD + dof]);
        }
        // V tile: vector loads then transpose-scatter into LDS
        Pack8 vreg[4]; int vr[4], vd[4];
#pragma unroll
        for (int i = 0; i < 4; i++) {
            const int c = tid + i * 128;
            vr[i] = c >> 3; vd[i] = (c & 7) * 8;
            vreg[i].q = *(const uint4*)&base[(size_t)(kb + vr[i]) * rowstride +
                                             2 * DD + h * HDD + vd[i]];
        }
#pragma unroll
        for (int i = 0; i < 4; i++)
#pragma unroll
            for (int e = 0; e < 8; e++)
                VsT[vd[i] + e][vr[i]] = vreg[i].u[e];
        wait_async0();
        __syncthreads();

        // S = Q K^T   (16 q rows x 64 keys per wave)
        v8f sf[4];
#pragma unroll
        for (int t = 0; t < 4; t++) {
            v8f a = vzero8();
#pragma unroll
            for (int kk = 0; kk < 64; kk += 32) {
                FragAB fa, fb;
                fa.q[0] = *(const uint4*)&Qs[wid * 16 + l16][kk + half * 8];
                fa.q[1] = *(const uint4*)&Qs[wid * 16 + l16][kk + 16 + half * 8];
                fb.q[0] = *(const uint4*)&Ks[t * 16 + l16][kk + half * 8];
                fb.q[1] = *(const uint4*)&Ks[t * 16 + l16][kk + 16 + half * 8];
                a = __builtin_amdgcn_wmma_f32_16x16x32_bf16(
                        false, fa.v, false, fb.v, (short)0, a, false, false);
            }
            sf[t] = a;
        }

        // scale + causal mask + online softmax
        float tmax[8];
#pragma unroll
        for (int v = 0; v < 8; v++) tmax[v] = -1e30f;
#pragma unroll
        for (int t = 0; t < 4; t++)
#pragma unroll
            for (int v = 0; v < 8; v++) {
                const int keyg = kb + t * 16 + l16;
                const int qg   = qbase + wid * 16 + v + 8 * half;
                float s = sf[t][v] * scale;
                if (keyg > qg) s = -1e30f;
                sf[t][v] = s;
                tmax[v] = fmaxf(tmax[v], s);
            }
#pragma unroll
        for (int v = 0; v < 8; v++)
#pragma unroll
            for (int m = 1; m < 16; m <<= 1)
                tmax[v] = fmaxf(tmax[v], __shfl_xor(tmax[v], m, 32));

        float alpha[8], psum[8];
#pragma unroll
        for (int v = 0; v < 8; v++) {
            float mnew = fmaxf(rm[v], tmax[v]);
            alpha[v] = __expf(rm[v] - mnew);
            rm[v] = mnew;
            psum[v] = 0.f;
        }
#pragma unroll
        for (int t = 0; t < 4; t++)
#pragma unroll
            for (int v = 0; v < 8; v++) {
                float p = __expf(sf[t][v] - rm[v]);
                sf[t][v] = p;
                psum[v] += p;
            }
#pragma unroll
        for (int v = 0; v < 8; v++) {
#pragma unroll
            for (int m = 1; m < 16; m <<= 1) psum[v] += __shfl_xor(psum[v], m, 32);
            rl[v] = rl[v] * alpha[v] + psum[v];
        }
#pragma unroll
        for (int t = 0; t < 4; t++)
#pragma unroll
            for (int v = 0; v < 8; v++) oacc[t][v] *= alpha[v];

        // P fragment (C layout) -> per-wave LDS -> A fragment layout
#pragma unroll
        for (int t = 0; t < 4; t++)
#pragma unroll
            for (int v = 0; v < 8; v++)
                Ps[wid][v + 8 * half][t * 16 + l16] = f2bf(sf[t][v]);

        // O += P V
#pragma unroll
        for (int t = 0; t < 4; t++) {
#pragma unroll
            for (int kk = 0; kk < 64; kk += 32) {
                FragAB fa, fb;
                fa.q[0] = *(const uint4*)&Ps[wid][l16][kk + half * 8];
                fa.q[1] = *(const uint4*)&Ps[wid][l16][kk + 16 + half * 8];
                fb.q[0] = *(const uint4*)&VsT[t * 16 + l16][kk + half * 8];
                fb.q[1] = *(const uint4*)&VsT[t * 16 + l16][kk + 16 + half * 8];
                oacc[t] = __builtin_amdgcn_wmma_f32_16x16x32_bf16(
                              false, fa.v, false, fb.v, (short)0, oacc[t], false, false);
            }
        }
        __syncthreads();
    }

    // Normalize and write (B, S, D) bf16: col = h*64 + ...
#pragma unroll
    for (int t = 0; t < 4; t++)
#pragma unroll
        for (int v = 0; v < 8; v++) {
            const int qg = qbase + wid * 16 + v + 8 * half;
            const int d  = h * HDD + t * 16 + l16;
            out[(size_t)(b * SD + qg) * DD + d] = f2bf(oacc[t][v] / rl[v]);
        }
}

// ---------------------------------------------------------------------------
// Final: LN + (D x 5) projection, one block per token
// ---------------------------------------------------------------------------
__global__ __launch_bounds__(256) void final_kernel(const float* __restrict__ hbuf,
                                                    const float* __restrict__ w,
                                                    const float* __restrict__ b,
                                                    const float* __restrict__ Wp,
                                                    const float* __restrict__ bp,
                                                    float* __restrict__ out) {
    const int row = blockIdx.x;
    const int tid = threadIdx.x;
    const int wid = tid >> 5, lane = tid & 31;
    __shared__ float xs[DD];
    __shared__ float sm_s[8], sm_q[8];
    __shared__ float red[8][WIND];

    const float* xr = hbuf + (size_t)row * DD;
    float vals[4], s = 0.f, sq = 0.f;
#pragma unroll
    for (int k = 0; k < 4; k++) {
        float v = xr[tid + 256 * k];
        vals[k] = v; s += v; sq += v * v;
    }
#pragma unroll
    for (int m = 1; m < 32; m <<= 1) {
        s  += __shfl_xor(s,  m, 32);
        sq += __shfl_xor(sq, m, 32);
    }
    if (lane == 0) { sm_s[wid] = s; sm_q[wid] = sq; }
    __syncthreads();
    float S = 0.f, Q = 0.f;
#pragma unroll
    for (int i = 0; i < 8; i++) { S += sm_s[i]; Q += sm_q[i]; }
    const float mean = S * (1.f / DD);
    const float rstd = rsqrtf(Q * (1.f / DD) - mean * mean + 1e-5f);
#pragma unroll
    for (int k = 0; k < 4; k++) {
        int c = tid + 256 * k;
        xs[c] = (vals[k] - mean) * rstd * w[c] + b[c];
    }
    __syncthreads();

    float accw[WIND];
#pragma unroll
    for (int wi = 0; wi < WIND; wi++) accw[wi] = 0.f;
    for (int d = tid; d < DD; d += 256) {
        float xv = xs[d];
#pragma unroll
        for (int wi = 0; wi < WIND; wi++) accw[wi] += xv * Wp[(size_t)d * WIND + wi];
    }
#pragma unroll
    for (int wi = 0; wi < WIND; wi++)
#pragma unroll
        for (int m = 1; m < 32; m <<= 1) accw[wi] += __shfl_xor(accw[wi], m, 32);
    if (lane == 0)
#pragma unroll
        for (int wi = 0; wi < WIND; wi++) red[wid][wi] = accw[wi];
    __syncthreads();
    if (tid < WIND) {
        float r = bp[tid];
#pragma unroll
        for (int i = 0; i < 8; i++) r += red[i][tid];
        out[(size_t)row * WIND + tid] = r;
    }
}

// ---------------------------------------------------------------------------
// Host orchestration
// ---------------------------------------------------------------------------
extern "C" void kernel_launch(void* const* d_in, const int* in_sizes, int n_in,
                              void* d_out, int out_size, void* d_ws, size_t ws_size,
                              hipStream_t stream) {
    (void)in_sizes; (void)n_in; (void)out_size; (void)ws_size;
    const float* x     = (const float*)d_in[0];
    const float* Wqkv  = (const float*)d_in[1];
    const float* bqkv  = (const float*)d_in[2];
    const float* Wo    = (const float*)d_in[3];
    const float* bo    = (const float*)d_in[4];
    const float* ln1w  = (const float*)d_in[5];
    const float* ln1b  = (const float*)d_in[6];
    const float* W1    = (const float*)d_in[7];
    const float* b1    = (const float*)d_in[8];
    const float* W2    = (const float*)d_in[9];
    const float* b2    = (const float*)d_in[10];
    const float* ln2w  = (const float*)d_in[11];
    const float* ln2b  = (const float*)d_in[12];
    const float* lnfw  = (const float*)d_in[13];
    const float* lnfb  = (const float*)d_in[14];
    const float* Wp    = (const float*)d_in[15];
    const float* bp    = (const float*)d_in[16];

    char* ws = (char*)d_ws;
    size_t off = 0;
    auto carve = [&](size_t bytes) { void* p = ws + off; off += (bytes + 255) & ~(size_t)255; return p; };
    float* hbuf  = (float*)carve((size_t)MTOK * DD * 4);          // hidden state
    u16*   hnb   = (u16*)  carve((size_t)MTOK * DD * 2);          // LN / attn-out bf16
    u16*   qkvb  = (u16*)  carve((size_t)MTOK * 3 * DD * 2);      // QKV bf16
    u16*   wb    = (u16*)  carve((size_t)DD * DFFD * 2);          // weight bf16 (transposed)
    u16*   ff1b  = (u16*)  carve((size_t)MTOK * DFFD * 2);        // FFN mid bf16

    hipMemcpyAsync(hbuf, x, (size_t)MTOK * DD * 4, hipMemcpyDeviceToDevice, stream);

    const dim3 gemm_blk(256);
    for (int l = 0; l < LD; l++) {
        // --- attention sub-block ---
        ln_bf16_kernel<<<MTOK, 256, 0, stream>>>(hbuf, ln1w + l * DD, ln1b + l * DD, hnb);
        cvtT_kernel<<<dim3(3 * DD / TT, DD / TT), 256, 0, stream>>>(
            Wqkv + (size_t)l * DD * 3 * DD, wb, DD, 3 * DD);
        gemm_bf16_kernel<<<dim3(3 * DD / TN, MTOK / TM), gemm_blk, 0, stream>>>(
            hnb, wb, bqkv + l * 3 * DD, nullptr, nullptr, qkvb, MTOK, 3 * DD, DD, 0);
        attn_kernel<<<dim3(SD / 64, HH, BD), 128, 0, stream>>>(qkvb, hnb);
        cvtT_kernel<<<dim3(DD / TT, DD / TT), 256, 0, stream>>>(
            Wo + (size_t)l * DD * DD, wb, DD, DD);
        gemm_bf16_kernel<<<dim3(DD / TN, MTOK / TM), gemm_blk, 0, stream>>>(
            hnb, wb, bo + l * DD, hbuf, hbuf, nullptr, MTOK, DD, DD, 0);
        // --- FFN sub-block ---
        ln_bf16_kernel<<<MTOK, 256, 0, stream>>>(hbuf, ln2w + l * DD, ln2b + l * DD, hnb);
        cvtT_kernel<<<dim3(DFFD / TT, DD / TT), 256, 0, stream>>>(
            W1 + (size_t)l * DD * DFFD, wb, DD, DFFD);
        gemm_bf16_kernel<<<dim3(DFFD / TN, MTOK / TM), gemm_blk, 0, stream>>>(
            hnb, wb, b1 + l * DFFD, nullptr, nullptr, ff1b, MTOK, DFFD, DD, 1);
        cvtT_kernel<<<dim3(DD / TT, DFFD / TT), 256, 0, stream>>>(
            W2 + (size_t)l * DFFD * DD, wb, DFFD, DD);
        gemm_bf16_kernel<<<dim3(DD / TN, MTOK / TM), gemm_blk, 0, stream>>>(
            ff1b, wb, b2 + l * DD, hbuf, hbuf, nullptr, MTOK, DD, DFFD, 0);
    }
    final_kernel<<<MTOK, 256, 0, stream>>>(hbuf, lnfw, lnfb, Wp, bp, (float*)d_out);
}
